// HeterogeneousGAT_61881888801205
// MI455X (gfx1250) — matrix operand order repaired and checked by reading, not compiled, not run on hardware
//
#include <hip/hip_runtime.h>
#include <math.h>

#define NOPS 131072L
#define NRES 16384L
#define NEDG 1048576L
#define NACT 4096L

typedef __attribute__((ext_vector_type(16))) __bf16 v16bf;
typedef __attribute__((ext_vector_type(8)))  __bf16 v8bf;
typedef __attribute__((ext_vector_type(8)))  float  v8f;

// branchless ELU: elu(x) = max(x, exp(min(x,0))-1)  (exp(x)-1 >= x for all x)
__device__ __forceinline__ float elu_fast(float x) {
  return fmaxf(x, __expf(fminf(x, 0.f)) - 1.f);
}

// ---------------- WMMA fragment helpers (ISA 7.12.2 layouts, wave32) ----------------

__device__ __forceinline__ v16bf frag2(const __bf16* p0, const __bf16* p1) {
  v8bf a = *(const v8bf*)p0;
  v8bf b = *(const v8bf*)p1;
  v16bf f;
#pragma unroll
  for (int i = 0; i < 8; ++i) { f[i] = a[i]; f[i + 8] = b[i]; }
  return f;
}

// B fragment (K x 16 tile) from global bf16 weights stored transposed row-major [N][Kq].
// lane: N = t*16 + (lane&15); base = 8*(lane>=16); h<8 -> K=k*32+base+h ; h>=8 -> K=k*32+base+16+(h-8)
__device__ __forceinline__ v16bf bfrag(const __bf16* Wt, int Kq, int t, int k, int lane) {
  int n  = t * 16 + (lane & 15);
  int kb = k * 32 + ((lane & 16) ? 8 : 0);
  const __bf16* p = Wt + (long)n * Kq + kb;
  return frag2(p, p + 16);
}

// A fragment (16 x 32 tile) from LDS bf16 [16][stride] row-major.
// lane: M = lane&15; base = 8*(lane>=16); h<8 -> K=k*32+base+h ; h>=8 -> K=k*32+base+16+(h-8)
__device__ __forceinline__ v16bf afrag_lds(const __bf16* h, int stride, int k, int lane) {
  int m = lane & 15;
  int base = (lane & 16) ? 8 : 0;
  const __bf16* p = h + m * stride + k * 32 + base;
  return frag2(p, p + 16);
}

template <int ACT>
__device__ __forceinline__ float actf(float x) {
  if (ACT == 0) return elu_fast(x);  // ELU, branchless
  return tanhf(x);                   // tanh
}

// ---------------- fused 3-layer MLP via v_wmma_f32_16x16x32_bf16 ----------------
// Y[M x OUT] = L3( act(L2( act(L1(X)) )) ); rows outside [lo,hi] are written as 0.
// NSRC==1: X0 row-major [M][IN_DIM] (with row shift, clamped)
// NSRC==4: columns are concat of X0..X3 each [M][8]
template <int IN_DIM, int HID, int OUT, int ACT, int NSRC>
__global__ __launch_bounds__(256) void mlp_wmma_kernel(
    const float* X0, const float* X1, const float* X2, const float* X3,
    int shift, long M, long lo, long hi,
    const unsigned short* w1t, const float* b1,
    const unsigned short* w2t, const float* b2,
    const unsigned short* w3t, const float* b3,
    float* Y) {
  constexpr int NT = HID / 16;   // N tiles
  constexpr int KS = HID / 32;   // K steps for layers 2/3
  constexpr int SH = HID + 8;    // padded LDS row stride (bank spread, 16B aligned)
  __shared__ __bf16 sH[8][16 * SH];

  const __bf16* W1T = (const __bf16*)w1t;
  const __bf16* W2T = (const __bf16*)w2t;
  const __bf16* W3T = (const __bf16*)w3t;

  int wave = threadIdx.x >> 5;
  int lane = threadIdx.x & 31;
  long row0 = (long)blockIdx.x * 128 + (long)wave * 16;
  __bf16* hbuf = &sH[wave][0];

  // ---- layer 1: A fragment gathered straight from global f32 (zero-padded K to 32)
  const float* Xs[4] = {X0, X1, X2, X3};
  long grow = row0 + (lane & 15) + shift;
  grow = grow < 0 ? 0 : (grow > M - 1 ? M - 1 : grow);
  int base = (lane & 16) ? 8 : 0;
  v16bf a1;
#pragma unroll
  for (int h = 0; h < 16; ++h) {
    int k = base + (h < 8 ? h : h + 8);
    float v = 0.f;
    if (k < IN_DIM) {
      if constexpr (NSRC == 1) v = X0[grow * IN_DIM + k];
      else                     v = Xs[k >> 3][grow * 8 + (k & 7)];
    }
    a1[h] = (__bf16)v;
  }
#pragma unroll
  for (int t = 0; t < NT; ++t) {
    v8f acc = {};
    v16bf b = bfrag(W1T, 32, t, 0, lane);
    acc = __builtin_amdgcn_wmma_f32_16x16x32_bf16(false, a1, false, b, (short)0, acc, false, false);
    int n = t * 16 + (lane & 15);
    float bias = b1[n];
    int mb = (lane & 16) ? 8 : 0;
#pragma unroll
    for (int r = 0; r < 8; ++r) hbuf[(r + mb) * SH + n] = (__bf16)actf<ACT>(acc[r] + bias);
  }

  // ---- layer 2: HID x HID
  v16bf a2[KS];
#pragma unroll
  for (int k = 0; k < KS; ++k) a2[k] = afrag_lds(hbuf, SH, k, lane);
#pragma unroll
  for (int t = 0; t < NT; ++t) {
    v8f acc = {};
#pragma unroll
    for (int k = 0; k < KS; ++k) {
      v16bf b = bfrag(W2T, HID, t, k, lane);
      acc = __builtin_amdgcn_wmma_f32_16x16x32_bf16(false, a2[k], false, b, (short)0, acc, false, false);
    }
    int n = t * 16 + (lane & 15);
    float bias = b2[n];
    int mb = (lane & 16) ? 8 : 0;
#pragma unroll
    for (int r = 0; r < 8; ++r) hbuf[(r + mb) * SH + n] = (__bf16)actf<ACT>(acc[r] + bias);
  }

  // ---- layer 3: HID -> OUT (single padded N tile)
  v8f acc = {};
#pragma unroll
  for (int k = 0; k < KS; ++k) {
    v16bf a3 = afrag_lds(hbuf, SH, k, lane);
    v16bf b  = bfrag(W3T, HID, 0, k, lane);
    acc = __builtin_amdgcn_wmma_f32_16x16x32_bf16(false, a3, false, b, (short)0, acc, false, false);
  }
  int n = lane & 15;
  if (n < OUT) {
    float bias = b3[n];
    int mb = (lane & 16) ? 8 : 0;
#pragma unroll
    for (int r = 0; r < 8; ++r) {
      long orow = row0 + r + mb;
      if (orow < M) {
        float v = (orow >= lo && orow <= hi) ? (acc[r] + bias) : 0.f;
        Y[orow * OUT + n] = v;
      }
    }
  }
}

// ---------------- weight prep: f32 [K][N] -> bf16 transposed zero-padded [Nq][Kq] ----------------
struct PrepJob { const float* src; unsigned short* dst; int K, N, Kq, Nq; };
struct PrepArgs { PrepJob j[33]; };

__global__ __launch_bounds__(256) void prep_weights_kernel(PrepArgs a) {
  PrepJob job = a.j[blockIdx.y];
  __bf16* dst = (__bf16*)job.dst;
  int total = job.Kq * job.Nq;
  for (int i = blockIdx.x * 256 + threadIdx.x; i < total; i += gridDim.x * 256) {
    int n = i / job.Kq, k = i - n * job.Kq;
    float v = (k < job.K && n < job.N) ? job.src[(long)k * job.N + n] : 0.f;
    dst[i] = (__bf16)v;
  }
}

// ---------------- resource-layer kernels ----------------

__global__ __launch_bounds__(256) void rproj_kernel(const float* R, int rd, const float* Wr,
    const float* aself, const float* avec, float* r_proj, float* as_s, float* rs_s, long n) {
  long i = (long)blockIdx.x * 256 + threadIdx.x;
  if (i >= n) return;
  float x[8];
  for (int k = 0; k < rd; ++k) x[k] = R[i * rd + k];
  float asv = 0.f, rsv = 0.f;
#pragma unroll
  for (int j = 0; j < 8; ++j) {
    float v = 0.f;
    for (int k = 0; k < rd; ++k) v += x[k] * Wr[k * 8 + j];
    r_proj[i * 8 + j] = v;
    asv += v * (aself[j] + aself[8 + j]);
    rsv += v * avec[j];
  }
  as_s[i] = asv > 0.f ? asv : 0.2f * asv;   // leaky_relu(0.2)
  rs_s[i] = rsv;
}

__global__ __launch_bounds__(256) void oproj_kernel(const float* O, int od, const float* Wo,
    const float* avec_hi, float* o_proj, float* os_s, long n) {
  long i = (long)blockIdx.x * 256 + threadIdx.x;
  if (i >= n) return;
  float x[8];
  for (int k = 0; k < od; ++k) x[k] = O[i * od + k];
  float osv = 0.f;
#pragma unroll
  for (int j = 0; j < 8; ++j) {
    float v = 0.f;
    for (int k = 0; k < od; ++k) v += x[k] * Wo[k * 8 + j];
    o_proj[i * 8 + j] = v;
    osv += v * avec_hi[j];
  }
  os_s[i] = osv;
}

__global__ __launch_bounds__(256) void edge_score_kernel(const int* req0, const int* req1,
    const float* rs_s, const float* os_s, float* a_cross, long n) {
  long e = (long)blockIdx.x * 256 + threadIdx.x;
  if (e >= n) return;
  float v = rs_s[req1[e]] + os_s[req0[e]];
  a_cross[e] = v > 0.f ? v : 0.2f * v;
}

// ---------------- global softmax (two-pass reductions over concat(as_s, a_cross)) ----------------

__global__ __launch_bounds__(256) void partial_max_kernel(const float* A, long nA,
    const float* B, long nB, float* part) {
  __shared__ float red[256];
  long n = nA + nB;
  float m = -1e30f;
  for (long i = (long)blockIdx.x * 256 + threadIdx.x; i < n; i += (long)gridDim.x * 256) {
    float v = i < nA ? A[i] : B[i - nA];
    m = fmaxf(m, v);
  }
  red[threadIdx.x] = m; __syncthreads();
  for (int s = 128; s > 0; s >>= 1) {
    if (threadIdx.x < s) red[threadIdx.x] = fmaxf(red[threadIdx.x], red[threadIdx.x + s]);
    __syncthreads();
  }
  if (threadIdx.x == 0) part[blockIdx.x] = red[0];
}

__global__ __launch_bounds__(256) void final_max_kernel(const float* part, int n, float* scal) {
  __shared__ float red[256];
  float m = -1e30f;
  for (int i = threadIdx.x; i < n; i += 256) m = fmaxf(m, part[i]);
  red[threadIdx.x] = m; __syncthreads();
  for (int s = 128; s > 0; s >>= 1) {
    if (threadIdx.x < s) red[threadIdx.x] = fmaxf(red[threadIdx.x], red[threadIdx.x + s]);
    __syncthreads();
  }
  if (threadIdx.x == 0) scal[0] = red[0];
}

__global__ __launch_bounds__(256) void partial_sum_kernel(const float* A, long nA,
    const float* B, long nB, const float* scal, float* part) {
  __shared__ float red[256];
  long n = nA + nB;
  float gm = scal[0];
  float s = 0.f;
  for (long i = (long)blockIdx.x * 256 + threadIdx.x; i < n; i += (long)gridDim.x * 256) {
    float v = i < nA ? A[i] : B[i - nA];
    s += __expf(v - gm);
  }
  red[threadIdx.x] = s; __syncthreads();
  for (int st = 128; st > 0; st >>= 1) {
    if (threadIdx.x < st) red[threadIdx.x] += red[threadIdx.x + st];
    __syncthreads();
  }
  if (threadIdx.x == 0) part[blockIdx.x] = red[0];
}

__global__ __launch_bounds__(256) void final_sum_kernel(const float* part, int n, float* scal) {
  __shared__ float red[256];
  float s = 0.f;
  for (int i = threadIdx.x; i < n; i += 256) s += part[i];
  red[threadIdx.x] = s; __syncthreads();
  for (int st = 128; st > 0; st >>= 1) {
    if (threadIdx.x < st) red[threadIdx.x] += red[threadIdx.x + st];
    __syncthreads();
  }
  if (threadIdx.x == 0) scal[1] = 1.f / red[0];  // store reciprocal
}

__global__ __launch_bounds__(256) void agg_edges_kernel(const int* req0, const int* req1,
    const float* a_cross, const float* o_proj, const float* scal, float* agg, long n) {
  long e = (long)blockIdx.x * 256 + threadIdx.x;
  if (e >= n) return;
  float w = __expf(a_cross[e] - scal[0]) * scal[1];
  const float* o = o_proj + (long)req0[e] * 8;
  float* a = agg + (long)req1[e] * 8;
#pragma unroll
  for (int j = 0; j < 8; ++j) atomicAdd(&a[j], w * o[j]);
}

__global__ __launch_bounds__(256) void res_update_kernel(const float* r_proj, const float* as_s,
    const float* scal, const float* agg, float* res_out, long n) {
  long i = (long)blockIdx.x * 256 + threadIdx.x;
  if (i >= n) return;
  float ns = __expf(as_s[i] - scal[0]) * scal[1];
#pragma unroll
  for (int j = 0; j < 8; ++j) {
    float v = ns * r_proj[i * 8 + j] + agg[i * 8 + j];
    res_out[i * 8 + j] = elu_fast(v);
  }
}

__global__ __launch_bounds__(256) void aggmach_kernel(const int* req0, const int* req1,
    const float* res_cur, float* agg_mach, long n) {
  long e = (long)blockIdx.x * 256 + threadIdx.x;
  if (e >= n) return;
  const float* r = res_cur + (long)req1[e] * 8;
  float* a = agg_mach + (long)req0[e] * 8;
#pragma unroll
  for (int j = 0; j < 8; ++j) atomicAdd(&a[j], r[j]);
}

// ---------------- tail: means, critic, actor gather, softmax ----------------

__global__ __launch_bounds__(256) void colmean8_kernel(const float* X, long n, float scale, float* out) {
  __shared__ float sm[256 * 8];
  float acc[8] = {0.f, 0.f, 0.f, 0.f, 0.f, 0.f, 0.f, 0.f};
  for (long i = (long)blockIdx.x * 256 + threadIdx.x; i < n; i += (long)gridDim.x * 256) {
    const float* row = X + i * 8;
#pragma unroll
    for (int j = 0; j < 8; ++j) acc[j] += row[j];
  }
#pragma unroll
  for (int j = 0; j < 8; ++j) sm[threadIdx.x * 8 + j] = acc[j];
  __syncthreads();
  for (int s = 128; s > 0; s >>= 1) {
    if (threadIdx.x < s)
#pragma unroll
      for (int j = 0; j < 8; ++j) sm[threadIdx.x * 8 + j] += sm[(threadIdx.x + s) * 8 + j];
    __syncthreads();
  }
  if (threadIdx.x < 8) atomicAdd(&out[threadIdx.x], sm[threadIdx.x] * scale);
}

__global__ void critic_kernel(const float* gsbuf,
    const float* w1, const float* b1, const float* w2, const float* b2,
    const float* w3, const float* b3, float* out) {
  __shared__ float gs[16], h1[64], h2[64];
  int t = threadIdx.x;
  if (t < 16) gs[t] = gsbuf[t];
  __syncthreads();
  float s = b1[t];
  for (int k = 0; k < 16; ++k) s += gs[k] * w1[k * 64 + t];
  h1[t] = tanhf(s);
  __syncthreads();
  s = b2[t];
  for (int k = 0; k < 64; ++k) s += h1[k] * w2[k * 64 + t];
  h2[t] = tanhf(s);
  __syncthreads();
  if (t == 0) {
    float v = b3[0];
    for (int k = 0; k < 64; ++k) v += h2[k] * w3[k];
    out[0] = v;
  }
}

__global__ __launch_bounds__(256) void act_gather_kernel(const int* acts, const float* ops_cur,
    const float* res_cur, const float* gsbuf, float* act_in, long n) {
  long i = (long)blockIdx.x * 256 + threadIdx.x;
  if (i >= n) return;
  long a0 = acts[2 * i], a1 = acts[2 * i + 1];
  float* o = act_in + i * 32;
#pragma unroll
  for (int j = 0; j < 8; ++j) { o[j] = ops_cur[a0 * 8 + j]; o[8 + j] = res_cur[a1 * 8 + j]; }
#pragma unroll
  for (int j = 0; j < 16; ++j) o[16 + j] = gsbuf[j];
}

__global__ __launch_bounds__(256) void softmax_kernel(const float* x, int n, float* out) {
  __shared__ float red[256];
  float m = -1e30f;
  for (int i = threadIdx.x; i < n; i += 256) m = fmaxf(m, x[i]);
  red[threadIdx.x] = m; __syncthreads();
  for (int s = 128; s > 0; s >>= 1) {
    if (threadIdx.x < s) red[threadIdx.x] = fmaxf(red[threadIdx.x], red[threadIdx.x + s]);
    __syncthreads();
  }
  float gm = red[0]; __syncthreads();
  float s = 0.f;
  for (int i = threadIdx.x; i < n; i += 256) s += __expf(x[i] - gm);
  red[threadIdx.x] = s; __syncthreads();
  for (int st = 128; st > 0; st >>= 1) {
    if (threadIdx.x < st) red[threadIdx.x] += red[threadIdx.x + st];
    __syncthreads();
  }
  float inv = 1.f / red[0];
  for (int i = threadIdx.x; i < n; i += 256) out[i] = __expf(x[i] - gm) * inv;
}

// ---------------- host launcher ----------------

extern "C" void kernel_launch(void* const* d_in, const int* in_sizes, int n_in,
                              void* d_out, int out_size, void* d_ws, size_t ws_size,
                              hipStream_t stream) {
  (void)in_sizes; (void)n_in; (void)out_size; (void)ws_size;
  const float* OPS0 = (const float*)d_in[0];
  const float* RES0 = (const float*)d_in[1];
  const int*   REQ  = (const int*)d_in[3];
  const int*   ACTS = (const int*)d_in[4];
  const int P = 6;  // params start (recursive dict-insertion order flattening)
  auto F = [&](int i) -> const float* { return (const float*)d_in[i]; };

  // workspace carve
  char* wsb = (char*)d_ws;
  size_t off = 0;
  auto alloc = [&](size_t bytes) -> void* {
    off = (off + 255) & ~(size_t)255;
    void* p = wsb + off;
    off += bytes;
    return p;
  };
  float* r_proj   = (float*)alloc(NRES * 8 * 4);
  float* o_proj   = (float*)alloc(NOPS * 8 * 4);
  float* as_s     = (float*)alloc(NRES * 4);
  float* rs_s     = (float*)alloc(NRES * 4);
  float* os_s     = (float*)alloc(NOPS * 4);
  float* a_cross  = (float*)alloc(NEDG * 4);
  float* agg      = (float*)alloc(NRES * 8 * 4);
  float* res_cur  = (float*)alloc(NRES * 8 * 4);
  float* ops_cur  = (float*)alloc(NOPS * 8 * 4);
  float* agg_mach = (float*)alloc(NOPS * 8 * 4);
  float* m_pred   = (float*)alloc(NOPS * 8 * 4);
  float* m_succ   = (float*)alloc(NOPS * 8 * 4);
  float* m_same   = (float*)alloc(NOPS * 8 * 4);
  float* m_aggm   = (float*)alloc(NOPS * 8 * 4);
  float* part     = (float*)alloc(1024 * 4);
  float* scal     = (float*)alloc(16 * 4);
  float* gsbuf    = (float*)alloc(16 * 4);
  float* act_in   = (float*)alloc(NACT * 32 * 4);
  float* logits   = (float*)alloc(NACT * 4);

  // bf16 weight staging: 10 op-MLPs + actor
  unsigned short *W1T[11], *W2T[11], *W3T[11];
  PrepArgs pa{};
  int jn = 0;
  auto add_mlp = [&](int pbase, int in_dim, int hid, int outd, int idx) {
    W1T[idx] = (unsigned short*)alloc((size_t)hid * 32 * 2);
    W2T[idx] = (unsigned short*)alloc((size_t)hid * hid * 2);
    W3T[idx] = (unsigned short*)alloc((size_t)16 * hid * 2);
    pa.j[jn++] = PrepJob{F(pbase + 0), W1T[idx], in_dim, hid, 32, hid};
    pa.j[jn++] = PrepJob{F(pbase + 2), W2T[idx], hid, hid, hid, hid};
    pa.j[jn++] = PrepJob{F(pbase + 4), W3T[idx], hid, outd, hid, 16};
  };
  for (int l = 0; l < 2; ++l) {
    int ob = P + 8 + l * 30;
    int ind = (l == 0) ? 6 : 8;
    add_mlp(ob + 0,  ind, 128, 8, l * 5 + 0);  // pred
    add_mlp(ob + 6,  ind, 128, 8, l * 5 + 1);  // succ
    add_mlp(ob + 12, ind, 128, 8, l * 5 + 2);  // same
    add_mlp(ob + 18, 8,   128, 8, l * 5 + 3);  // res
    add_mlp(ob + 24, 32,  128, 8, l * 5 + 4);  // comb
  }
  add_mlp(P + 68, 32, 64, 1, 10);              // actor
  prep_weights_kernel<<<dim3(64, jn), 256, 0, stream>>>(pa);

  const float* Osrc = OPS0; int od = 6;
  const float* Rsrc = RES0; int rd = 3;
  for (int l = 0; l < 2; ++l) {
    int rb = P + l * 4;  // Wr, Wo, a_self, a
    // --- resource layer ---
    rproj_kernel<<<NRES / 256, 256, 0, stream>>>(Rsrc, rd, F(rb + 0), F(rb + 2), F(rb + 3),
                                                 r_proj, as_s, rs_s, NRES);
    oproj_kernel<<<NOPS / 256, 256, 0, stream>>>(Osrc, od, F(rb + 1), F(rb + 3) + 8,
                                                 o_proj, os_s, NOPS);
    edge_score_kernel<<<NEDG / 256, 256, 0, stream>>>(REQ, REQ + NEDG, rs_s, os_s, a_cross, NEDG);
    partial_max_kernel<<<1024, 256, 0, stream>>>(as_s, NRES, a_cross, NEDG, part);
    final_max_kernel<<<1, 256, 0, stream>>>(part, 1024, scal);
    partial_sum_kernel<<<1024, 256, 0, stream>>>(as_s, NRES, a_cross, NEDG, scal, part);
    final_sum_kernel<<<1, 256, 0, stream>>>(part, 1024, scal);
    hipMemsetAsync(agg, 0, NRES * 8 * 4, stream);
    agg_edges_kernel<<<NEDG / 256, 256, 0, stream>>>(REQ, REQ + NEDG, a_cross, o_proj, scal, agg, NEDG);
    res_update_kernel<<<NRES / 256, 256, 0, stream>>>(r_proj, as_s, scal, agg, res_cur, NRES);
    // --- operation layer ---
    hipMemsetAsync(agg_mach, 0, NOPS * 8 * 4, stream);
    aggmach_kernel<<<NEDG / 256, 256, 0, stream>>>(REQ, REQ + NEDG, res_cur, agg_mach, NEDG);
    int mi = l * 5;
#define MB(i) (P + 8 + l * 30 + (i) * 6)
    dim3 g((unsigned)(NOPS / 128));
    if (l == 0) {
      mlp_wmma_kernel<6, 128, 8, 0, 1><<<g, 256, 0, stream>>>(Osrc, nullptr, nullptr, nullptr, -1,
          NOPS, 1, NOPS - 2, W1T[mi + 0], F(MB(0) + 1), W2T[mi + 0], F(MB(0) + 3), W3T[mi + 0], F(MB(0) + 5), m_pred);
      mlp_wmma_kernel<6, 128, 8, 0, 1><<<g, 256, 0, stream>>>(Osrc, nullptr, nullptr, nullptr, +1,
          NOPS, 1, NOPS - 2, W1T[mi + 1], F(MB(1) + 1), W2T[mi + 1], F(MB(1) + 3), W3T[mi + 1], F(MB(1) + 5), m_succ);
      mlp_wmma_kernel<6, 128, 8, 0, 1><<<g, 256, 0, stream>>>(Osrc, nullptr, nullptr, nullptr, 0,
          NOPS, 1, NOPS - 2, W1T[mi + 2], F(MB(2) + 1), W2T[mi + 2], F(MB(2) + 3), W3T[mi + 2], F(MB(2) + 5), m_same);
    } else {
      mlp_wmma_kernel<8, 128, 8, 0, 1><<<g, 256, 0, stream>>>(Osrc, nullptr, nullptr, nullptr, -1,
          NOPS, 1, NOPS - 2, W1T[mi + 0], F(MB(0) + 1), W2T[mi + 0], F(MB(0) + 3), W3T[mi + 0], F(MB(0) + 5), m_pred);
      mlp_wmma_kernel<8, 128, 8, 0, 1><<<g, 256, 0, stream>>>(Osrc, nullptr, nullptr, nullptr, +1,
          NOPS, 1, NOPS - 2, W1T[mi + 1], F(MB(1) + 1), W2T[mi + 1], F(MB(1) + 3), W3T[mi + 1], F(MB(1) + 5), m_succ);
      mlp_wmma_kernel<8, 128, 8, 0, 1><<<g, 256, 0, stream>>>(Osrc, nullptr, nullptr, nullptr, 0,
          NOPS, 1, NOPS - 2, W1T[mi + 2], F(MB(2) + 1), W2T[mi + 2], F(MB(2) + 3), W3T[mi + 2], F(MB(2) + 5), m_same);
    }
    mlp_wmma_kernel<8, 128, 8, 0, 1><<<g, 256, 0, stream>>>(agg_mach, nullptr, nullptr, nullptr, 0,
        NOPS, 1, NOPS - 2, W1T[mi + 3], F(MB(3) + 1), W2T[mi + 3], F(MB(3) + 3), W3T[mi + 3], F(MB(3) + 5), m_aggm);
    // comb: concat order [preds, succs, aggm, same]
    mlp_wmma_kernel<32, 128, 8, 0, 4><<<g, 256, 0, stream>>>(m_pred, m_succ, m_aggm, m_same, 0,
        NOPS, 1, NOPS - 2, W1T[mi + 4], F(MB(4) + 1), W2T[mi + 4], F(MB(4) + 3), W3T[mi + 4], F(MB(4) + 5), ops_cur);
#undef MB
    Osrc = ops_cur; od = 8;
    Rsrc = res_cur; rd = 8;
  }

  // --- graph state, critic, actor, softmax ---
  hipMemsetAsync(gsbuf, 0, 16 * 4, stream);
  colmean8_kernel<<<256, 256, 0, stream>>>(ops_cur, NOPS, 1.f / (float)NOPS, gsbuf);
  colmean8_kernel<<<256, 256, 0, stream>>>(res_cur, NRES, 1.f / (float)NRES, gsbuf + 8);
  int cb = P + 74;
  critic_kernel<<<1, 64, 0, stream>>>(gsbuf, F(cb + 0), F(cb + 1), F(cb + 2), F(cb + 3),
                                      F(cb + 4), F(cb + 5), (float*)d_out + 4096);
  act_gather_kernel<<<NACT / 256, 256, 0, stream>>>(ACTS, ops_cur, res_cur, gsbuf, act_in, NACT);
  int ab = P + 68;
  mlp_wmma_kernel<32, 64, 1, 1, 1><<<dim3((unsigned)(NACT / 128)), 256, 0, stream>>>(
      act_in, nullptr, nullptr, nullptr, 0, NACT, 0, NACT - 1,
      W1T[10], F(ab + 1), W2T[10], F(ab + 3), W3T[10], F(ab + 5), logits);
  softmax_kernel<<<1, 256, 0, stream>>>(logits, (int)NACT, (float*)d_out);
}
